// GRU_Modified_43258910605996
// MI455X (gfx1250) — matrix operand order repaired and checked by reading, not compiled, not run on hardware
//
#include <hip/hip_runtime.h>
#include <hip/hip_bf16.h>

#define B_  64
#define T_  512
#define I_  512
#define H_  1024
#define XH_ 1536          // I_ + H_
#define KT_ (XH_ / 32)    // 48 k-steps of 32
#define ROWSTR 1544       // padded LDS row stride (772 dwords ≡ 4 mod 64 banks)

typedef __attribute__((ext_vector_type(16))) __bf16 bf16x16;
typedef __attribute__((ext_vector_type(8)))  __bf16 bf16x8;
typedef __attribute__((ext_vector_type(8)))  float  f32x8;
typedef __attribute__((ext_vector_type(4)))  int    i32x4;

// ---------------------------------------------------------------------------
// Activations: prefer gfx1250 native v_tanh_f32; else v_rcp_f32 (no IEEE div).
// ---------------------------------------------------------------------------
#if __has_builtin(__builtin_amdgcn_tanhf)
__device__ __forceinline__ float tanh_(float x) {
    return __builtin_amdgcn_tanhf(x);
}
__device__ __forceinline__ float sigmoid_(float x) {
    return 0.5f + 0.5f * __builtin_amdgcn_tanhf(0.5f * x);
}
#else
__device__ __forceinline__ float tanh_(float x) {
    return 1.0f - 2.0f * __builtin_amdgcn_rcpf(1.0f + __expf(2.0f * x));
}
__device__ __forceinline__ float sigmoid_(float x) {
    return __builtin_amdgcn_rcpf(1.0f + __expf(-x));
}
#endif

// ---------------------------------------------------------------------------
// Async global->LDS staging (CDNA5), with a synchronous fallback.
// Builtin signature (from hipcc diagnostic): arg0 = v4i32 addrspace(1)*,
// then LDS pointer, imm offset, imm cpol.
// ---------------------------------------------------------------------------
#if __has_builtin(__builtin_amdgcn_global_load_async_to_lds_b128)
#define ASYNC_LDS 1
#endif

__device__ __forceinline__ void copy16_to_lds(const __bf16* g, __bf16* l) {
#ifdef ASYNC_LDS
    __builtin_amdgcn_global_load_async_to_lds_b128(
        (__attribute__((address_space(1))) i32x4*)g,
        (__attribute__((address_space(3))) i32x4*)l, 0, 0);
#else
    *(bf16x8*)l = *(const bf16x8*)g;
#endif
}

__device__ __forceinline__ void wait_staging() {
#ifdef ASYNC_LDS
#if __has_builtin(__builtin_amdgcn_s_wait_asynccnt)
    __builtin_amdgcn_s_wait_asynccnt(0);
#else
    asm volatile("s_wait_asynccnt 0x0" ::: "memory");
#endif
#endif
    __syncthreads();
}

// ---------------------------------------------------------------------------
// fp32 -> bf16 elementwise conversion (grid-stride)
// ---------------------------------------------------------------------------
__global__ void f32_to_bf16_kernel(const float* __restrict__ src,
                                   __bf16* __restrict__ dst, int n) {
    int i = blockIdx.x * blockDim.x + threadIdx.x;
    int stride = gridDim.x * blockDim.x;
    for (; i < n; i += stride) dst[i] = (__bf16)src[i];
}

#define WMMA_BF16(a, b, c) \
    __builtin_amdgcn_wmma_f32_16x16x32_bf16(false, (a), false, (b), (short)0, \
                                            (c), false, false)

// A fragment from the LDS panel (16x32 layout):
//   lane L -> row M = L&15; elems 0..15 = K half*16+0..15
#define LOAD_A_LDS(a_, kt_)                                                   \
    {                                                                         \
        bf16x8 lo = *(const bf16x8*)(arow + (kt_)*32);                        \
        bf16x8 hi = *(const bf16x8*)(arow + (kt_)*32 + 8);                    \
        _Pragma("unroll") for (int e = 0; e < 8; ++e) {                       \
            (a_)[e] = lo[e];                                                  \
            (a_)[8 + e] = hi[e];                                              \
        }                                                                     \
    }

// ---------------------------------------------------------------------------
// Kernel 1: rz = [x_t, h] @ W_rz^T + b_rz ; r=sig(rz_r); z=sig(rz_z + 1)
// 4 waves/block share one 16-row A panel staged async into LDS.
// Each wave owns a 16x64 strip (4 accumulators); B loads double-buffered.
// Strips: (B/16)*(2H/64) = 128 waves -> 32 blocks x 128 threads.
// ---------------------------------------------------------------------------
__global__ void __launch_bounds__(128)
gru_rz_step(const __bf16* __restrict__ xbf,   // [B,T,I] bf16
            const __bf16* __restrict__ hbf,   // [B,H]   bf16
            const float*  __restrict__ hf,    // [B,H]   fp32
            const __bf16* __restrict__ Wrz,   // [2H,XH] bf16 row-major
            const float*  __restrict__ brz,   // [2H]
            __bf16* __restrict__ rh,          // [B,H]   bf16 out
            float*  __restrict__ zf,          // [B,H]   fp32 out
            int t) {
    __shared__ __bf16 panel[16 * ROWSTR];     // [row][K] of [x_t, h], 48.25KB
    const int tid  = threadIdx.x;
    const int widx = tid >> 5;
    const int lane = tid & 31;
    const int half = lane >> 4;
    const int lrow = lane & 15;
    const int NSTRIPS = (2 * H_) / 64;        // 32
    const int wave0  = blockIdx.x * 4;
    const int mBase  = (wave0 / NSTRIPS) * 16;            // same for all waves
    const int nBase0 = ((wave0 + widx) % NSTRIPS) * 64;

    // ---- stage A panel: x rows (64 x 16B chunks/row), h rows (128/row) ----
#pragma unroll
    for (int it = 0; it < 8; ++it) {          // 1024 x-chunks / 128 threads
        const int c = tid + it * 128;
        const int row = c >> 6, col = (c & 63) * 8;
        copy16_to_lds(xbf + (((size_t)(mBase + row)) * T_ + t) * I_ + col,
                      &panel[row * ROWSTR + col]);
    }
#pragma unroll
    for (int it = 0; it < 16; ++it) {         // 2048 h-chunks / 128 threads
        const int c = tid + it * 128;
        const int row = c >> 7, col = (c & 127) * 8;
        copy16_to_lds(hbf + (size_t)(mBase + row) * H_ + col,
                      &panel[row * ROWSTR + I_ + col]);
    }
    wait_staging();

    const __bf16* arow = &panel[lrow * ROWSTR + half * 16];
    const __bf16* W0 = Wrz + (size_t)(nBase0 + 0  + lrow) * XH_ + half * 16;
    const __bf16* W1 = Wrz + (size_t)(nBase0 + 16 + lrow) * XH_ + half * 16;
    const __bf16* W2 = Wrz + (size_t)(nBase0 + 32 + lrow) * XH_ + half * 16;
    const __bf16* W3 = Wrz + (size_t)(nBase0 + 48 + lrow) * XH_ + half * 16;

    f32x8 acc0 = {}, acc1 = {}, acc2 = {}, acc3 = {};
    bf16x16 b0c = *(const bf16x16*)(W0);
    bf16x16 b1c = *(const bf16x16*)(W1);
    bf16x16 b2c = *(const bf16x16*)(W2);
    bf16x16 b3c = *(const bf16x16*)(W3);
#pragma unroll 2
    for (int kt = 0; kt < KT_ - 1; ++kt) {    // B double-buffered
        const int kN = (kt + 1) * 32;
        bf16x16 b0n = *(const bf16x16*)(W0 + kN);
        bf16x16 b1n = *(const bf16x16*)(W1 + kN);
        bf16x16 b2n = *(const bf16x16*)(W2 + kN);
        bf16x16 b3n = *(const bf16x16*)(W3 + kN);
        bf16x16 a;
        LOAD_A_LDS(a, kt);
        acc0 = WMMA_BF16(a, b0c, acc0);
        acc1 = WMMA_BF16(a, b1c, acc1);
        acc2 = WMMA_BF16(a, b2c, acc2);
        acc3 = WMMA_BF16(a, b3c, acc3);
        b0c = b0n; b1c = b1n; b2c = b2n; b3c = b3n;
    }
    {
        bf16x16 a;
        LOAD_A_LDS(a, KT_ - 1);
        acc0 = WMMA_BF16(a, b0c, acc0);
        acc1 = WMMA_BF16(a, b1c, acc1);
        acc2 = WMMA_BF16(a, b2c, acc2);
        acc3 = WMMA_BF16(a, b3c, acc3);
    }

    // C layout: VGPR j -> (M = half*8 + j, N = lane&15)
    f32x8 accs[4] = {acc0, acc1, acc2, acc3};
#pragma unroll
    for (int i = 0; i < 4; ++i) {
        const int n = nBase0 + i * 16 + lrow;
        if (n < H_) {                              // r gate (uniform per strip)
            const float bias = brz[n];
#pragma unroll
            for (int j = 0; j < 8; ++j) {
                const int m = mBase + half * 8 + j;
                float r = sigmoid_(accs[i][j] + bias);
                rh[(size_t)m * H_ + n] = (__bf16)(r * hf[(size_t)m * H_ + n]);
            }
        } else {                                   // z gate (+UPDATE_BIAS)
            const float bias = brz[n] + 1.0f;
            const int nz = n - H_;
#pragma unroll
            for (int j = 0; j < 8; ++j) {
                const int m = mBase + half * 8 + j;
                zf[(size_t)m * H_ + nz] = sigmoid_(accs[i][j] + bias);
            }
        }
    }
}

// ---------------------------------------------------------------------------
// Kernel 2: hhat = tanh([x_t, r*h] @ W_h^T + b_h); h_new = z*h + (1-z)*hhat
// Strips: (B/16)*(H/64) = 64 waves -> 16 blocks x 128 threads.
// ---------------------------------------------------------------------------
__global__ void __launch_bounds__(128)
gru_h_step(const __bf16* __restrict__ xbf,    // [B,T,I] bf16
           const __bf16* __restrict__ rhbf,   // [B,H]   bf16 (r*h)
           float*  __restrict__ hf,           // [B,H]   fp32 (in/out)
           const float* __restrict__ zf,      // [B,H]   fp32
           const __bf16* __restrict__ Wh,     // [H,XH]  bf16 row-major
           const float*  __restrict__ bh,     // [H]
           __bf16* __restrict__ hbf,          // [B,H]   bf16 (out, next step)
           float*  __restrict__ out,          // [B,T,H] fp32
           int t) {
    __shared__ __bf16 panel[16 * ROWSTR];     // [row][K] of [x_t, r*h]
    const int tid  = threadIdx.x;
    const int widx = tid >> 5;
    const int lane = tid & 31;
    const int half = lane >> 4;
    const int lrow = lane & 15;
    const int NSTRIPS = H_ / 64;              // 16
    const int wave0  = blockIdx.x * 4;
    const int mBase  = (wave0 / NSTRIPS) * 16;
    const int nBase0 = ((wave0 + widx) % NSTRIPS) * 64;

#pragma unroll
    for (int it = 0; it < 8; ++it) {
        const int c = tid + it * 128;
        const int row = c >> 6, col = (c & 63) * 8;
        copy16_to_lds(xbf + (((size_t)(mBase + row)) * T_ + t) * I_ + col,
                      &panel[row * ROWSTR + col]);
    }
#pragma unroll
    for (int it = 0; it < 16; ++it) {
        const int c = tid + it * 128;
        const int row = c >> 7, col = (c & 127) * 8;
        copy16_to_lds(rhbf + (size_t)(mBase + row) * H_ + col,
                      &panel[row * ROWSTR + I_ + col]);
    }
    wait_staging();

    const __bf16* arow = &panel[lrow * ROWSTR + half * 16];
    const __bf16* W0 = Wh + (size_t)(nBase0 + 0  + lrow) * XH_ + half * 16;
    const __bf16* W1 = Wh + (size_t)(nBase0 + 16 + lrow) * XH_ + half * 16;
    const __bf16* W2 = Wh + (size_t)(nBase0 + 32 + lrow) * XH_ + half * 16;
    const __bf16* W3 = Wh + (size_t)(nBase0 + 48 + lrow) * XH_ + half * 16;

    f32x8 acc0 = {}, acc1 = {}, acc2 = {}, acc3 = {};
    bf16x16 b0c = *(const bf16x16*)(W0);
    bf16x16 b1c = *(const bf16x16*)(W1);
    bf16x16 b2c = *(const bf16x16*)(W2);
    bf16x16 b3c = *(const bf16x16*)(W3);
#pragma unroll 2
    for (int kt = 0; kt < KT_ - 1; ++kt) {
        const int kN = (kt + 1) * 32;
        bf16x16 b0n = *(const bf16x16*)(W0 + kN);
        bf16x16 b1n = *(const bf16x16*)(W1 + kN);
        bf16x16 b2n = *(const bf16x16*)(W2 + kN);
        bf16x16 b3n = *(const bf16x16*)(W3 + kN);
        bf16x16 a;
        LOAD_A_LDS(a, kt);
        acc0 = WMMA_BF16(a, b0c, acc0);
        acc1 = WMMA_BF16(a, b1c, acc1);
        acc2 = WMMA_BF16(a, b2c, acc2);
        acc3 = WMMA_BF16(a, b3c, acc3);
        b0c = b0n; b1c = b1n; b2c = b2n; b3c = b3n;
    }
    {
        bf16x16 a;
        LOAD_A_LDS(a, KT_ - 1);
        acc0 = WMMA_BF16(a, b0c, acc0);
        acc1 = WMMA_BF16(a, b1c, acc1);
        acc2 = WMMA_BF16(a, b2c, acc2);
        acc3 = WMMA_BF16(a, b3c, acc3);
    }

    f32x8 accs[4] = {acc0, acc1, acc2, acc3};
#pragma unroll
    for (int i = 0; i < 4; ++i) {
        const int n = nBase0 + i * 16 + lrow;
        const float bias = bh[n];
#pragma unroll
        for (int j = 0; j < 8; ++j) {
            const int m  = mBase + half * 8 + j;
            const size_t hidx = (size_t)m * H_ + n;
            float hhat  = tanh_(accs[i][j] + bias);
            float z     = zf[hidx];
            float hprev = hf[hidx];
            float hnew  = z * hprev + (1.0f - z) * hhat;
            hf[hidx]  = hnew;
            hbf[hidx] = (__bf16)hnew;
            // write-once output: keep it out of L2 so weights stay resident
            __builtin_nontemporal_store(hnew,
                                        out + ((size_t)m * T_ + t) * H_ + n);
        }
    }
}

// ---------------------------------------------------------------------------
// Host side: prep conversions once, then 512 (rz, h) step pairs on `stream`.
// Workspace: Wrz(6.3MB) + Wh(3.1MB) + x(33.6MB) + state buffers (~0.8MB).
// ---------------------------------------------------------------------------
extern "C" void kernel_launch(void* const* d_in, const int* in_sizes, int n_in,
                              void* d_out, int out_size, void* d_ws,
                              size_t ws_size, hipStream_t stream) {
    const float* x   = (const float*)d_in[0];   // [B,T,I]
    const float* h0  = (const float*)d_in[1];   // [B,H]
    const float* Wrz = (const float*)d_in[2];   // [2H,XH]
    const float* brz = (const float*)d_in[3];   // [2H]
    const float* Wh  = (const float*)d_in[4];   // [H,XH]
    const float* bh  = (const float*)d_in[5];   // [H]
    float* out = (float*)d_out;

    char* ws = (char*)d_ws;
    size_t off = 0;
    auto take = [&](size_t bytes) -> void* {
        off = (off + 255) & ~(size_t)255;
        void* p = ws + off;
        off += bytes;
        return p;
    };

    __bf16* Wrz_bf = (__bf16*)take((size_t)2 * H_ * XH_ * 2);
    __bf16* Wh_bf  = (__bf16*)take((size_t)H_ * XH_ * 2);
    __bf16* x_bf   = (__bf16*)take((size_t)B_ * T_ * I_ * 2);
    float*  h_f    = (float*)take((size_t)B_ * H_ * 4);
    __bf16* h_bf   = (__bf16*)take((size_t)B_ * H_ * 2);
    __bf16* rh_bf  = (__bf16*)take((size_t)B_ * H_ * 2);
    float*  z_f    = (float*)take((size_t)B_ * H_ * 4);
    (void)ws_size; (void)in_sizes; (void)n_in; (void)out_size;

    f32_to_bf16_kernel<<<1024, 256, 0, stream>>>(Wrz, Wrz_bf, 2 * H_ * XH_);
    f32_to_bf16_kernel<<<512,  256, 0, stream>>>(Wh,  Wh_bf,  H_ * XH_);
    f32_to_bf16_kernel<<<4096, 256, 0, stream>>>(x,   x_bf,   B_ * T_ * I_);
    f32_to_bf16_kernel<<<64,   256, 0, stream>>>(h0,  h_bf,   B_ * H_);
    (void)hipMemcpyAsync(h_f, h0, (size_t)B_ * H_ * sizeof(float),
                         hipMemcpyDeviceToDevice, stream);

    for (int t = 0; t < T_; ++t) {
        gru_rz_step<<<32, 128, 0, stream>>>(x_bf, h_bf, h_f, Wrz_bf, brz,
                                            rh_bf, z_f, t);
        gru_h_step<<<16, 128, 0, stream>>>(x_bf, rh_bf, h_f, z_f, Wh_bf, bh,
                                           h_bf, out, t);
    }
}
